// DeformConv_21199958573533
// MI455X (gfx1250) — matrix operand order repaired and checked by reading, not compiled, not run on hardware
//
#include <hip/hip_runtime.h>
#include <hip/hip_bf16.h>
#include <stdint.h>

typedef __attribute__((ext_vector_type(16))) __bf16 v16bf;
typedef __attribute__((ext_vector_type(8)))  float  v8f;

#define B_    4
#define C_    256
#define H_    96
#define W_    96
#define HW_   9216
#define NP_   36864      /* B*H*W  */
#define CHO_  256
#define NPC_  9437184    /* NP_*C_ */

/* workspace layout (bytes) */
#define XB_OFF     0u           /* NHWC bf16 x: NP_*C_*2 = 18,874,368 */
#define WOFF_OFF   18874368u    /* w_off_t [9][32][256] bf16 = 147,456 */
#define WT_OFF     19021824u    /* weight_t [9][256][256] bf16 = 1,179,648 */
#define OFFBUF_OFF 20201472u    /* 27 planes * NP_ fp32 = 3,981,312 */
#define STATS_OFF  24182784u    /* 512 fp32 (sum, sumsq) */

/* ---------------- prep: layout transforms + zero stats ---------------- */
__global__ __launch_bounds__(256) void prep_kernel(
    const float* __restrict__ x, const float* __restrict__ w_off,
    const float* __restrict__ weight,
    __bf16* __restrict__ xb, __bf16* __restrict__ wofft,
    __bf16* __restrict__ wt, float* __restrict__ stats)
{
    int idx = blockIdx.x * 256 + threadIdx.x;
    if (idx < NPC_) {                      /* x NCHW f32 -> NHWC bf16 */
        int c = idx & 255;
        int p = idx >> 8;
        int b = p / HW_;
        int hw = p - b * HW_;
        xb[idx] = (__bf16)x[((size_t)(b * C_ + c)) * HW_ + hw];
    }
    if (idx < 9 * 32 * 256) {              /* w_off [27][C][3][3] -> [t][n(32)][c] */
        int c = idx & 255;
        int n = (idx >> 8) & 31;
        int t = idx >> 13;
        float v = (n < 27) ? w_off[((size_t)(n * C_ + c)) * 9 + t] : 0.0f;
        wofft[idx] = (__bf16)v;
    }
    if (idx < 9 * 256 * 256) {             /* weight [O][C][3][3] -> [t][o][c] */
        int c = idx & 255;
        int o = (idx >> 8) & 255;
        int t = idx >> 16;
        wt[idx] = (__bf16)weight[((size_t)(o * C_ + c)) * 9 + t];
    }
    if (idx < 512) stats[idx] = 0.0f;
}

/* ------------- offset/mask conv: WMMA implicit GEMM, N=32 ------------- */
__global__ __launch_bounds__(256) void offconv_kernel(
    const __bf16* __restrict__ xb, const __bf16* __restrict__ wofft,
    const float* __restrict__ b_off, float* __restrict__ offbuf)
{
    __shared__ __align__(16) __bf16 As[128][40];   /* 128 px x 32 K (+pad) */
    __shared__ __align__(16) __bf16 Bs[32][40];    /* 32 n  x 32 K (+pad) */

    const int tid  = threadIdx.x;
    const int wave = tid >> 5;
    const int lane = tid & 31;
    const int pbase = blockIdx.x * 128;

    v8f acc0 = {}; v8f acc1 = {};

    const int ar   = tid >> 1;
    const int ac16 = (tid & 1) * 16;
    const int ap = pbase + ar;
    const int ab = ap / HW_;
    const int ahw = ap - ab * HW_;
    const int ay = ahw / W_;
    const int ax = ahw - ay * W_;

    for (int t = 0; t < 9; ++t) {
        const int ky = t / 3, kx = t - ky * 3;
        const int yy = ay + ky - 1;
        const int xx = ax + kx - 1;
        const bool valid = (yy >= 0) & (yy < H_) & (xx >= 0) & (xx < W_);
        const __bf16* src = xb + ((size_t)((ab * H_ + yy) * W_ + xx)) * C_;
        for (int cc = 0; cc < 8; ++cc) {
            const int c0 = cc * 32;
            __syncthreads();
            uint4 q0 = make_uint4(0, 0, 0, 0), q1 = q0;
            if (valid) {
                const uint4* sp = (const uint4*)(src + c0 + ac16);
                q0 = sp[0]; q1 = sp[1];
            }
            *(uint4*)&As[ar][ac16]     = q0;
            *(uint4*)&As[ar][ac16 + 8] = q1;
            {
                int n  = tid >> 3;
                int co = (tid & 7) * 4;
                *(uint2*)&Bs[n][co] =
                    *(const uint2*)(wofft + ((size_t)(t * 32 + n)) * 256 + c0 + co);
            }
            __syncthreads();
            const int khalf = lane >> 4;
            const int l15   = lane & 15;
            union AF { uint4 q[2]; v16bf v; } af, bf0, bf1;
            af.q[0]  = *(const uint4*)&As[wave * 16 + l15][16 * khalf];
            af.q[1]  = *(const uint4*)&As[wave * 16 + l15][16 * khalf + 8];
            bf0.q[0] = *(const uint4*)&Bs[l15][16 * khalf];
            bf0.q[1] = *(const uint4*)&Bs[l15][16 * khalf + 8];
            bf1.q[0] = *(const uint4*)&Bs[16 + l15][16 * khalf];
            bf1.q[1] = *(const uint4*)&Bs[16 + l15][16 * khalf + 8];
            acc0 = __builtin_amdgcn_wmma_f32_16x16x32_bf16(false, af.v, false, bf0.v,
                                                           (short)0, acc0, false, false);
            acc1 = __builtin_amdgcn_wmma_f32_16x16x32_bf16(false, af.v, false, bf1.v,
                                                           (short)0, acc1, false, false);
        }
    }
    const int khalf = lane >> 4;
    const int l15   = lane & 15;
    const int p0 = pbase + wave * 16 + 8 * khalf;
    #pragma unroll
    for (int nb = 0; nb < 2; ++nb) {
        int n = nb * 16 + l15;
        if (n < 27) {
            const v8f a = nb ? acc1 : acc0;
            float bo = b_off[n];
            float vals[8];
            #pragma unroll
            for (int r = 0; r < 8; ++r) {
                float v = a[r] + bo;
                if (n >= 18) v = 1.0f / (1.0f + __expf(-v));  /* sigmoid(mask) */
                vals[r] = v;
            }
            float4* dst = (float4*)(offbuf + (size_t)n * NP_ + p0);
            dst[0] = make_float4(vals[0], vals[1], vals[2], vals[3]);
            dst[1] = make_float4(vals[4], vals[5], vals[6], vals[7]);
        }
    }
}

/* -------- fused deformable sampling + WMMA GEMM + bias + BN stats -----
 * N-complete tile: 64 pixels x ALL 256 output channels per workgroup, so
 * every pixel is bilinear-sampled exactly once (4x less sampling VALU).
 * Per wave per K-step: 1 A-frag, 8 B-frags, 8 v_wmma (8 accumulators).
 * Double-buffered LDS, one barrier per K-step; B streamed with
 * GLOBAL_LOAD_ASYNC_TO_LDS_B128 (ASYNCcnt); A staged via registers.    */
__global__ __launch_bounds__(256) void dcn_kernel(
    const __bf16* __restrict__ xb, const __bf16* __restrict__ wt,
    const float* __restrict__ offbuf, const float* __restrict__ bias,
    float* __restrict__ out, float* __restrict__ gsum)
{
    __shared__ __align__(16) __bf16 As[2][64][40];    /* 64 px  x 32 ch, x2 */
    __shared__ __align__(16) __bf16 Bs[2][256][40];   /* 256 o  x 32 ch, x2 */
    __shared__ float ssum[256];
    __shared__ float ssum2[256];

    const int tid  = threadIdx.x;
    const int wave = tid >> 5;
    const int lane = tid & 31;
    const int wm = wave & 3;        /* M sub-tile (16 rows)   */
    const int wn = wave >> 2;       /* N half (128 channels)  */
    const int pbase = blockIdx.x * 64;

    ssum[tid] = 0.0f; ssum2[tid] = 0.0f;

    v8f acc[8];
    #pragma unroll
    for (int i = 0; i < 8; ++i) acc[i] = (v8f){};

    /* A-fill role: 4 threads per pixel row, each owns 8 channels */
    const int ar = tid >> 2;
    const int aj = (tid & 3) * 8;
    const int p  = pbase + ar;
    const int b  = p / HW_;
    const int hw = p - b * HW_;
    const int y  = hw / W_;
    const int x  = hw - y * W_;
    const int base = b * H_ * W_;

    float w0, w1, w2, w3;
    int   a0, a1, a2, a3;
    auto sampf = [&](int t) {
        const int ky = t / 3, kx = t - ky * 3;
        const float o1 = offbuf[(size_t)t * NP_ + p];
        const float o2 = offbuf[(size_t)(9 + t) * NP_ + p];
        const float mk = offbuf[(size_t)(18 + t) * NP_ + p];
        const float py = o1 + (float)(y + ky - 1);
        const float px = o2 + (float)(x + kx - 1);
        const float fy = floorf(py), fx = floorf(px);
        const float ly = py - fy, lx = px - fx;
        const int iy0 = (int)fy, ix0 = (int)fx;
        const int iy1 = iy0 + 1, ix1 = ix0 + 1;
        const bool vy0 = (iy0 >= 0) & (iy0 < H_);
        const bool vy1 = (iy1 >= 0) & (iy1 < H_);
        const bool vx0 = (ix0 >= 0) & (ix0 < W_);
        const bool vx1 = (ix1 >= 0) & (ix1 < W_);
        w0 = (vy0 & vx0) ? (1.f - ly) * (1.f - lx) * mk : 0.f;
        a0 = (vy0 & vx0) ? (base + iy0 * W_ + ix0) * C_ : 0;
        w1 = (vy0 & vx1) ? (1.f - ly) * lx * mk : 0.f;
        a1 = (vy0 & vx1) ? (base + iy0 * W_ + ix1) * C_ : 0;
        w2 = (vy1 & vx0) ? ly * (1.f - lx) * mk : 0.f;
        a2 = (vy1 & vx0) ? (base + iy1 * W_ + ix0) * C_ : 0;
        w3 = (vy1 & vx1) ? ly * lx * mk : 0.f;
        a3 = (vy1 & vx1) ? (base + iy1 * W_ + ix1) * C_ : 0;
    };

    union U8 { uint4 q; __bf16 h[8]; } u0, u1, u2, u3;
    auto loadA = [&](int step) {
        const int c0 = (step & 7) * 32;
        u0.q = *(const uint4*)(xb + (size_t)a0 + c0 + aj);
        u1.q = *(const uint4*)(xb + (size_t)a1 + c0 + aj);
        u2.q = *(const uint4*)(xb + (size_t)a2 + c0 + aj);
        u3.q = *(const uint4*)(xb + (size_t)a3 + c0 + aj);
    };

    /* B tile: thread `tid` streams one full weight row (32 ch = 64 B)
       with four async 16-byte global->LDS copies (ASYNCcnt-tracked). */
    auto issueB = [&](int step, int buf) {
        const int t  = step >> 3;
        const int c0 = (step & 7) * 32;
        const __bf16* gsrc = wt + ((size_t)(t * 256 + tid)) * 256 + c0;
        const unsigned l0 = (unsigned)(uintptr_t)&Bs[buf][tid][0];
        #pragma unroll
        for (int i = 0; i < 4; ++i) {
            asm volatile("global_load_async_to_lds_b128 %0, %1, off"
                         :: "v"(l0 + 16u * i), "v"(gsrc + 8 * i) : "memory");
        }
    };

    /* pipeline prologue */
    sampf(0);
    loadA(0);
    issueB(0, 0);

    for (int step = 0; step < 72; ++step) {
        const int buf = step & 1;
        /* compute + store this step's A tile from staged registers */
        {
            union U8 r;
            #pragma unroll
            for (int j = 0; j < 8; ++j) {
                float v = w0 * (float)u0.h[j] + w1 * (float)u1.h[j]
                        + w2 * (float)u2.h[j] + w3 * (float)u3.h[j];
                r.h[j] = (__bf16)v;
            }
            *(uint4*)&As[buf][ar][aj] = r.q;
        }
        /* stage next step's A gathers (overlap with barrier + WMMAs) */
        if (step < 71) {
            const int ns = step + 1;
            if ((ns & 7) == 0) sampf(ns >> 3);
            loadA(ns);
        }
        /* this buf's async B must have landed before anyone reads it */
        asm volatile("s_wait_asynccnt 0x0" ::: "memory");
        __syncthreads();
        /* prefetch next B tile into the other buffer (buf^1 readers all
           finished before the barrier we just passed) */
        if (step < 71) issueB(step + 1, buf ^ 1);

        const int khalf = lane >> 4;
        const int l15   = lane & 15;
        union AF { uint4 q[2]; v16bf v; } af;
        af.q[0] = *(const uint4*)&As[buf][wm * 16 + l15][16 * khalf];
        af.q[1] = *(const uint4*)&As[buf][wm * 16 + l15][16 * khalf + 8];
        #pragma unroll
        for (int nb = 0; nb < 8; ++nb) {
            union AF bf;
            const int nrow = wn * 128 + nb * 16 + l15;
            bf.q[0] = *(const uint4*)&Bs[buf][nrow][16 * khalf];
            bf.q[1] = *(const uint4*)&Bs[buf][nrow][16 * khalf + 8];
            acc[nb] = __builtin_amdgcn_wmma_f32_16x16x32_bf16(false, af.v, false, bf.v,
                                                              (short)0, acc[nb], false, false);
        }
    }

    /* epilogue: bias, vectorized NCHW store, BN partial sums */
    const int khalf = lane >> 4;
    const int l15   = lane & 15;
    const int p0 = pbase + wm * 16 + 8 * khalf;   /* 8 contiguous pixels/lane */
    const int ob = p0 / HW_;
    const int hw0 = p0 - ob * HW_;
    #pragma unroll
    for (int nb = 0; nb < 8; ++nb) {
        const int o = wn * 128 + nb * 16 + l15;
        const float bv = bias[o];
        float s = 0.f, s2 = 0.f;
        float vals[8];
        #pragma unroll
        for (int r = 0; r < 8; ++r) {
            float v = acc[nb][r] + bv;
            vals[r] = v;
            s += v; s2 += v * v;
        }
        float4* dst = (float4*)(out + ((size_t)(ob * CHO_ + o)) * HW_ + hw0);
        dst[0] = make_float4(vals[0], vals[1], vals[2], vals[3]);
        dst[1] = make_float4(vals[4], vals[5], vals[6], vals[7]);
        atomicAdd(&ssum[o], s);
        atomicAdd(&ssum2[o], s2);
    }
    __syncthreads();
    atomicAdd(&gsum[tid], ssum[tid]);
    atomicAdd(&gsum[256 + tid], ssum2[tid]);
}

/* --------------------- BN (batch stats) + ReLU, in place -------------- */
__global__ __launch_bounds__(256) void bn_relu_kernel(
    float* __restrict__ out, const float* __restrict__ gsum,
    const float* __restrict__ gamma, const float* __restrict__ beta)
{
    const int idx = blockIdx.x * 256 + threadIdx.x;
    const int o = (idx / HW_) & 255;
    const float inv = 1.0f / 36864.0f;
    const float mean = gsum[o] * inv;
    const float var  = gsum[256 + o] * inv - mean * mean;
    const float scale = gamma[o] * rsqrtf(var + 1e-5f);
    const float shift = beta[o] - mean * scale;
    float v = out[idx] * scale + shift;
    out[idx] = v > 0.f ? v : 0.f;
}

extern "C" void kernel_launch(void* const* d_in, const int* in_sizes, int n_in,
                              void* d_out, int out_size, void* d_ws, size_t ws_size,
                              hipStream_t stream)
{
    const float* x      = (const float*)d_in[0];
    const float* w_off  = (const float*)d_in[1];
    const float* b_off  = (const float*)d_in[2];
    const float* weight = (const float*)d_in[3];
    const float* bias   = (const float*)d_in[4];
    const float* gamma  = (const float*)d_in[5];
    const float* beta   = (const float*)d_in[6];

    char* ws = (char*)d_ws;
    __bf16* xb     = (__bf16*)(ws + XB_OFF);
    __bf16* wofft  = (__bf16*)(ws + WOFF_OFF);
    __bf16* wt     = (__bf16*)(ws + WT_OFF);
    float*  offbuf = (float*)(ws + OFFBUF_OFF);
    float*  stats  = (float*)(ws + STATS_OFF);
    float*  out    = (float*)d_out;

    prep_kernel<<<NPC_ / 256, 256, 0, stream>>>(x, w_off, weight, xb, wofft, wt, stats);
    offconv_kernel<<<NP_ / 128, 256, 0, stream>>>(xb, wofft, b_off, offbuf);
    dcn_kernel<<<NP_ / 64, 256, 0, stream>>>(xb, wt, offbuf, bias, out, stats);
    bn_relu_kernel<<<(NP_ * CHO_) / 256, 256, 0, stream>>>(out, stats, gamma, beta);
}
